// SwinTransformerBlock_90649579749905
// MI455X (gfx1250) — matrix operand order repaired and checked by reading, not compiled, not run on hardware
//
#include <hip/hip_runtime.h>
#include <cstdint>
#include <cstddef>

// ---------------- CDNA5 wave32 WMMA types ----------------
typedef __attribute__((ext_vector_type(16))) __bf16 bf16x16;
typedef __attribute__((ext_vector_type(8)))  __bf16 bf16x8;
typedef __attribute__((ext_vector_type(8)))  float  floatx8;

union ABFrag  { bf16x16 v; bf16x8 h[2]; };
union AccFrag { floatx8 v; float  e[8]; };

__device__ __forceinline__ floatx8 wmma_bf16(bf16x16 a, bf16x16 b, floatx8 c) {
  // D = A(16x32) * B(32x16) + C, f32 accumulate -> v_wmma_f32_16x16x32_bf16
  return __builtin_amdgcn_wmma_f32_16x16x32_bf16(
      false, a, false, b, (short)0, c, false, false);
}

// A fragment: 16x32 bf16, row-major source, leading dim `ld` (elements).
// lanes 0-15: row M=lane, K {0..7,16..23}; lanes 16-31: row M=lane-16, K {8..15,24..31}
__device__ __forceinline__ bf16x16 load_frag_a(const __bf16* base, int ld, int m0, int k0) {
  int lane = threadIdx.x & 31;
  int row  = m0 + (lane & 15);
  int kk   = k0 + ((lane & 16) ? 8 : 0);
  const __bf16* p = base + (size_t)row * ld + kk;
  ABFrag f;
  f.h[0] = *(const bf16x8*)(p);
  f.h[1] = *(const bf16x8*)(p + 16);
  return f.v;
}

// B fragment: B[k][n] = src[n][k], src row-major [N][K] leading dim ld.
// lanes 0-15: col N=lane, K {0..7,16..23}; lanes 16-31: col N=lane-16, K {8..15,24..31}
__device__ __forceinline__ bf16x16 load_frag_b(const __bf16* base, int ld, int n0, int k0) {
  int lane = threadIdx.x & 31;
  int col  = n0 + (lane & 15);
  int kk   = k0 + ((lane & 16) ? 8 : 0);
  const __bf16* p = base + (size_t)col * ld + kk;
  ABFrag f;
  f.h[0] = *(const bf16x8*)(p);
  f.h[1] = *(const bf16x8*)(p + 16);
  return f.v;
}

// ---------------- problem constants ----------------
#define BB   64
#define LL   2688
#define CC   256
#define WW   42
#define SS   21
#define HH   8
#define HD   32
#define NW   64
#define WP   48            // W padded to multiple of 16
#define KP   64            // attention K padded to multiple of 32
#define MLP  1024
#define MROWS (BB*LL)      // 172032

// ---------------- prep: f32 [K][N] -> bf16 [N][K] ----------------
__global__ __launch_bounds__(256) void k_transpose_cvt(const float* __restrict__ Win,
                                                       __bf16* __restrict__ Wt,
                                                       int K, int N, int total) {
  int idx = blockIdx.x * 256 + threadIdx.x;
  if (idx >= total) return;
  int k = idx / N, n = idx % N;
  Wt[(size_t)n * K + k] = (__bf16)Win[idx];
}

// zero the padded rows t=42..47 of q/k/v (runs every launch; ws is not re-zeroed)
__global__ __launch_bounds__(256) void k_zero_pad(__bf16* q, __bf16* k, __bf16* v, int total) {
  int idx = blockIdx.x * 256 + threadIdx.x;
  if (idx >= total) return;
  int hd = idx & 31;
  int t  = 42 + ((idx >> 5) % 6);
  int bh = idx / (6 * 32);
  size_t off = ((size_t)bh * WP + t) * HD + hd;
  q[off] = (__bf16)0.0f; k[off] = (__bf16)0.0f; v[off] = (__bf16)0.0f;
}

// ---------------- LayerNorm (one 256-thread block per token row) ----------------
// roll!=0: output row r maps to input position (r%L + S) mod L  (cyclic shift fused)
__global__ __launch_bounds__(256) void k_layernorm(const float* __restrict__ in,
                                                   const float* __restrict__ g,
                                                   const float* __restrict__ b,
                                                   __bf16* __restrict__ out, int roll) {
  __shared__ float ps[8];
  __shared__ float pq[8];
  int tid = threadIdx.x, wave = tid >> 5, lane = tid & 31;
  int row = blockIdx.x;
  int src;
  if (roll) {
    int bb  = row / LL;
    int rem = row % LL;
    src = bb * LL + ((rem + SS) % LL);
  } else {
    src = row;
  }
  float v = in[(size_t)src * CC + tid];
  float s = v, q = v * v;
#pragma unroll
  for (int o = 16; o > 0; o >>= 1) {
    s += __shfl_xor(s, o, 32);
    q += __shfl_xor(q, o, 32);
  }
  if (lane == 0) { ps[wave] = s; pq[wave] = q; }
  __syncthreads();
  float S = 0.f, Q = 0.f;
#pragma unroll
  for (int i = 0; i < 8; ++i) { S += ps[i]; Q += pq[i]; }
  float mean = S * (1.0f / CC);
  float var  = Q * (1.0f / CC) - mean * mean;
  float nv   = (v - mean) * rsqrtf(var + 1e-5f) * g[tid] + b[tid];
  out[(size_t)row * CC + tid] = (__bf16)nv;
}

// ---------------- generic WMMA GEMM: block = 256 rows x 64 cols, 8 waves ----------------
// Wave owns 32 rows x 64 cols (2 A-frags, 8 accumulators). K-loop is software-pipelined:
// fragments for step ks+1 are issued before the WMMAs of step ks consume their operands.
#define EPI_QKV  0
#define EPI_PROJ 1
#define EPI_FC1  2
#define EPI_FC2  3

template <int EPI>
__global__ __launch_bounds__(256) void k_gemm_bf16(
    const __bf16* __restrict__ A,    // [M][K] bf16 row-major
    const __bf16* __restrict__ Wt,   // [N][K] bf16 (pre-transposed weights)
    const float*  __restrict__ bias, // [N]
    int K,
    float*  __restrict__ fout,
    __bf16* __restrict__ bout,
    const float* __restrict__ resid,
    __bf16* __restrict__ qdst, __bf16* __restrict__ kdst, __bf16* __restrict__ vdst) {
  int wave = threadIdx.x >> 5;
  int lane = threadIdx.x & 31;
  int m0 = blockIdx.x * 256 + wave * 32;
  int n0 = blockIdx.y * 64;
  int N  = (int)(gridDim.y << 6);

  AccFrag acc[2][4];
#pragma unroll
  for (int mt = 0; mt < 2; ++mt)
#pragma unroll
    for (int nt = 0; nt < 4; ++nt) acc[mt][nt].v = (floatx8){};

  const int ksteps = K >> 5;
  // prologue loads (k-step 0)
  bf16x16 a0 = load_frag_a(A, K, m0,      0);
  bf16x16 a1 = load_frag_a(A, K, m0 + 16, 0);
  bf16x16 b0 = load_frag_b(Wt, K, n0,      0);
  bf16x16 b1 = load_frag_b(Wt, K, n0 + 16, 0);
  bf16x16 b2 = load_frag_b(Wt, K, n0 + 32, 0);
  bf16x16 b3 = load_frag_b(Wt, K, n0 + 48, 0);

  for (int ks = 0; ks < ksteps; ++ks) {
    bf16x16 ca0 = a0, ca1 = a1, cb0 = b0, cb1 = b1, cb2 = b2, cb3 = b3;
    if (ks + 1 < ksteps) {
      int kn = (ks + 1) << 5;
      if (lane == 0) {
        __builtin_prefetch(A + (size_t)m0 * K + kn + 32, 0, 1);        // k-step +2
        __builtin_prefetch(A + (size_t)(m0 + 16) * K + kn + 32, 0, 1);
      }
      a0 = load_frag_a(A, K, m0,      kn);
      a1 = load_frag_a(A, K, m0 + 16, kn);
      b0 = load_frag_b(Wt, K, n0,      kn);
      b1 = load_frag_b(Wt, K, n0 + 16, kn);
      b2 = load_frag_b(Wt, K, n0 + 32, kn);
      b3 = load_frag_b(Wt, K, n0 + 48, kn);
    }
    acc[0][0].v = wmma_bf16(ca0, cb0, acc[0][0].v);
    acc[1][0].v = wmma_bf16(ca1, cb0, acc[1][0].v);
    acc[0][1].v = wmma_bf16(ca0, cb1, acc[0][1].v);
    acc[1][1].v = wmma_bf16(ca1, cb1, acc[1][1].v);
    acc[0][2].v = wmma_bf16(ca0, cb2, acc[0][2].v);
    acc[1][2].v = wmma_bf16(ca1, cb2, acc[1][2].v);
    acc[0][3].v = wmma_bf16(ca0, cb3, acc[0][3].v);
    acc[1][3].v = wmma_bf16(ca1, cb3, acc[1][3].v);
  }

#pragma unroll
  for (int mt = 0; mt < 2; ++mt) {
#pragma unroll
    for (int nt = 0; nt < 4; ++nt) {
      int nbase = n0 + nt * 16;           // uniform per tile
      int n  = nbase + (lane & 15);
      float bv = bias[n];
      if constexpr (EPI == EPI_QKV) {
        // uniform per tile: destination tensor + head (no per-lane branching)
        int which = nbase >> 8;
        int head  = (nbase >> 5) & 7;
        __bf16* dst = (which == 0) ? qdst : ((which == 1) ? kdst : vdst);
        int hd = (nbase & 31) + (lane & 15);
#pragma unroll
        for (int rr = 0; rr < 8; ++rr) {
          int m = m0 + mt * 16 + rr + ((lane & 16) ? 8 : 0);
          int bwin = m / WW, t = m % WW;
          size_t off = (((size_t)bwin * HH + head) * WP + t) * HD + hd;
          dst[off] = (__bf16)(acc[mt][nt].e[rr] + bv);
        }
      } else {
#pragma unroll
        for (int rr = 0; rr < 8; ++rr) {
          int m = m0 + mt * 16 + rr + ((lane & 16) ? 8 : 0);
          float val = acc[mt][nt].e[rr] + bv;
          if constexpr (EPI == EPI_PROJ) {
            // reverse cyclic shift + residual: x1[b, (rem+S)%L, n] = x[...] + val
            int bb = m / LL, rem = m % LL;
            int l  = (rem + SS) % LL;
            size_t idx = ((size_t)bb * LL + l) * CC + n;
            fout[idx] = resid[idx] + val;
          } else if constexpr (EPI == EPI_FC1) {
            float gv = 0.5f * val * (1.0f + erff(val * 0.70710678118654752f)); // exact GELU
            bout[(size_t)m * N + n] = (__bf16)gv;
          } else { // EPI_FC2
            size_t idx = (size_t)m * CC + n;
            fout[idx] = resid[idx] + val;
          }
        }
      }
    }
  }
}

// ---------------- windowed attention: one block per (window, head) ----------------
__global__ __launch_bounds__(256) void k_attention(
    const __bf16* __restrict__ Q, const __bf16* __restrict__ Kk,
    const __bf16* __restrict__ V, const float* __restrict__ bias_table,
    __bf16* __restrict__ obf) {
  __shared__ __align__(16) float  sS[WP][WP];   // prefilled additive matrix, then scores
  __shared__ __align__(16) __bf16 sP[WP][KP];   // probs  48x64 bf16 (K padded)
  __shared__ __align__(16) __bf16 sVt[HD][KP];  // V^T    32x64 bf16
  __shared__ float sBias[2 * WW - 1];           // rel-pos bias column for this head (83)

  int tid = threadIdx.x, wave = tid >> 5, lane = tid & 31;
  int bh  = blockIdx.x;
  int bwin = bh >> 3, h = bh & 7;
  const __bf16* q = Q  + (size_t)bh * (WP * HD);
  const __bf16* k = Kk + (size_t)bh * (WP * HD);
  const __bf16* v = V  + (size_t)bh * (WP * HD);

  // stage rel-pos bias column (83 floats) and V^T (zero K-pad cols 48..63)
  if (tid < 2 * WW - 1) sBias[tid] = bias_table[tid * HH + h];
  for (int idx = tid; idx < HD * KP; idx += 256) {
    int n = idx >> 6, kk = idx & 63;
    sVt[n][kk] = (kk < WP) ? v[(size_t)kk * HD + n] : (__bf16)0.0f;
  }
  __syncthreads();

  // prefill sS with additive term: rel-pos bias + shift mask, -1e30 outside the window
  bool maskOn = ((bwin & (NW - 1)) == NW - 1);
  for (int idx = tid; idx < WP * WP; idx += 256) {
    int m = idx / WP, n = idx % WP;
    float a = -1e30f;
    if (m < WW && n < WW) {
      a = sBias[n - m + WW - 1];
      if (maskOn && ((m < SS) != (n < SS))) a -= 100.0f;
    }
    sS[m][n] = a;
  }
  __syncthreads();

  // phase 1: sS += scale * (Q K^T)  (3x3 tiles, 1 WMMA each; epilogue is pure fma)
  const float scale = 0.17677669529663687f;  // HD^-0.5
  for (int t = wave; t < 9; t += 8) {
    int tm = t / 3, tn = t % 3;
    bf16x16 a = load_frag_a(q, HD, tm * 16, 0);
    bf16x16 b = load_frag_b(k, HD, tn * 16, 0);   // B[k][n] = K[n][k] -> Q K^T
    floatx8 c = {};
    c = wmma_bf16(a, b, c);
    AccFrag cf; cf.v = c;
#pragma unroll
    for (int rr = 0; rr < 8; ++rr) {
      int m = tm * 16 + rr + ((lane & 16) ? 8 : 0);
      int n = tn * 16 + (lane & 15);
      sS[m][n] = fmaf(cf.e[rr], scale, sS[m][n]);   // each element owned by one lane
    }
  }
  __syncthreads();

  // phase 2: softmax, one wave per row; lane c covers cols {c, c+32}
  for (int m = wave; m < WP; m += 8) {
    float v0 = sS[m][lane];
    float v1 = (lane < 16) ? sS[m][32 + lane] : -1e30f;
    float mx = fmaxf(v0, v1);
#pragma unroll
    for (int o = 16; o > 0; o >>= 1) mx = fmaxf(mx, __shfl_xor(mx, o, 32));
    float e0 = __expf(v0 - mx);
    float e1 = (lane < 16) ? __expf(v1 - mx) : 0.0f;
    float s  = e0 + e1;
#pragma unroll
    for (int o = 16; o > 0; o >>= 1) s += __shfl_xor(s, o, 32);
    float inv = (s > 0.0f) ? (1.0f / s) : 0.0f;
    sP[m][lane]      = (__bf16)(e0 * inv);
    sP[m][32 + lane] = (lane < 16) ? (__bf16)(e1 * inv) : (__bf16)0.0f; // zeros K-pad 48..63
  }
  __syncthreads();

  // phase 3: O = P V (3 M-tiles x 2 N-tiles, K=64 -> 2 WMMA per tile), LDS-fed
  for (int t = wave; t < 6; t += 8) {
    int tm = t >> 1, tn = t & 1;
    floatx8 c = {};
#pragma unroll
    for (int ks = 0; ks < 2; ++ks) {
      bf16x16 a = load_frag_a(&sP[0][0],  KP, tm * 16, ks * 32);
      bf16x16 b = load_frag_b(&sVt[0][0], KP, tn * 16, ks * 32);
      c = wmma_bf16(a, b, c);
    }
    AccFrag cf; cf.v = c;
#pragma unroll
    for (int rr = 0; rr < 8; ++rr) {
      int m = tm * 16 + rr + ((lane & 16) ? 8 : 0);
      if (m < WW) {
        int n = h * HD + tn * 16 + (lane & 15);
        obf[((size_t)bwin * WW + m) * CC + n] = (__bf16)cf.e[rr];
      }
    }
  }
}

// ---------------- host: orchestrate on `stream`, scratch from d_ws ----------------
extern "C" void kernel_launch(void* const* d_in, const int* in_sizes, int n_in,
                              void* d_out, int out_size, void* d_ws, size_t ws_size,
                              hipStream_t stream) {
  (void)in_sizes; (void)n_in; (void)out_size; (void)ws_size;
  const float* x        = (const float*)d_in[0];
  const float* norm1_g  = (const float*)d_in[1];
  const float* norm1_b  = (const float*)d_in[2];
  const float* qkv_w    = (const float*)d_in[3];
  const float* qkv_b    = (const float*)d_in[4];
  const float* btab     = (const float*)d_in[5];
  const float* proj_w   = (const float*)d_in[6];
  const float* proj_b   = (const float*)d_in[7];
  const float* norm2_g  = (const float*)d_in[8];
  const float* norm2_b  = (const float*)d_in[9];
  const float* fc1_w    = (const float*)d_in[10];
  const float* fc1_b    = (const float*)d_in[11];
  const float* fc2_w    = (const float*)d_in[12];
  const float* fc2_b    = (const float*)d_in[13];
  float* out = (float*)d_out;

  uint8_t* ws = (uint8_t*)d_ws;
  size_t off = 0;
  auto carve = [&](size_t bytes) -> void* {
    void* p = ws + off;
    off += (bytes + 255) & ~(size_t)255;
    return p;
  };
  __bf16* hbf    = (__bf16*)carve((size_t)MROWS * CC * 2);            // LN1 output (shifted order)
  __bf16* qbuf   = (__bf16*)carve((size_t)BB*NW*HH * WP * HD * 2);    // padded Q
  __bf16* kbuf   = (__bf16*)carve((size_t)BB*NW*HH * WP * HD * 2);    // padded K
  __bf16* vbuf   = (__bf16*)carve((size_t)BB*NW*HH * WP * HD * 2);    // padded V
  __bf16* obf    = (__bf16*)carve((size_t)MROWS * CC * 2);            // attention out (heads merged)
  float*  x1f    = (float*) carve((size_t)MROWS * CC * 4);            // residual-1 result
  __bf16* h2bf   = (__bf16*)carve((size_t)MROWS * CC * 2);            // LN2 output
  __bf16* m1bf   = (__bf16*)carve((size_t)MROWS * MLP * 2);           // GELU(fc1) output
  __bf16* qkvWt  = (__bf16*)carve((size_t)3*CC * CC * 2);             // [768][256]
  __bf16* projWt = (__bf16*)carve((size_t)CC * CC * 2);               // [256][256]
  __bf16* fc1Wt  = (__bf16*)carve((size_t)MLP * CC * 2);              // [1024][256]
  __bf16* fc2Wt  = (__bf16*)carve((size_t)CC * MLP * 2);              // [256][1024]

  // weights -> bf16 [N][K]
  k_transpose_cvt<<<(CC*3*CC + 255)/256, 256, 0, stream>>>(qkv_w,  qkvWt,  CC,  3*CC, CC*3*CC);
  k_transpose_cvt<<<(CC*CC   + 255)/256, 256, 0, stream>>>(proj_w, projWt, CC,  CC,   CC*CC);
  k_transpose_cvt<<<(CC*MLP  + 255)/256, 256, 0, stream>>>(fc1_w,  fc1Wt,  CC,  MLP,  CC*MLP);
  k_transpose_cvt<<<(MLP*CC  + 255)/256, 256, 0, stream>>>(fc2_w,  fc2Wt,  MLP, CC,   MLP*CC);

  // zero the 6 pad rows of each (window,head) q/k/v slab
  {
    int total = BB*NW*HH * 6 * HD;
    k_zero_pad<<<(total + 255)/256, 256, 0, stream>>>(qbuf, kbuf, vbuf, total);
  }

  // LN1 + fused cyclic shift
  k_layernorm<<<MROWS, 256, 0, stream>>>(x, norm1_g, norm1_b, hbf, 1);

  // QKV GEMM: 172032 x 768 x 256, scatter to padded q/k/v
  k_gemm_bf16<EPI_QKV><<<dim3(MROWS/256, (3*CC)/64), 256, 0, stream>>>(
      hbf, qkvWt, qkv_b, CC, nullptr, nullptr, nullptr, qbuf, kbuf, vbuf);

  // windowed attention, one block per (window, head)
  k_attention<<<BB*NW*HH, 256, 0, stream>>>(qbuf, kbuf, vbuf, btab, obf);

  // proj GEMM + reverse shift + residual -> x1
  k_gemm_bf16<EPI_PROJ><<<dim3(MROWS/256, CC/64), 256, 0, stream>>>(
      obf, projWt, proj_b, CC, x1f, nullptr, x, nullptr, nullptr, nullptr);

  // LN2
  k_layernorm<<<MROWS, 256, 0, stream>>>(x1f, norm2_g, norm2_b, h2bf, 0);

  // FC1 + GELU
  k_gemm_bf16<EPI_FC1><<<dim3(MROWS/256, MLP/64), 256, 0, stream>>>(
      h2bf, fc1Wt, fc1_b, CC, nullptr, m1bf, nullptr, nullptr, nullptr, nullptr);

  // FC2 + residual -> out
  k_gemm_bf16<EPI_FC2><<<dim3(MROWS/256, CC/64), 256, 0, stream>>>(
      m1bf, fc2Wt, fc2_b, MLP, out, nullptr, x1f, nullptr, nullptr, nullptr);
}
